// AudioEncoder_86457691669014
// MI455X (gfx1250) — matrix operand order repaired and checked by reading, not compiled, not run on hardware
//
#include <hip/hip_runtime.h>

// ---------------------------------------------------------------------------
// CDNA5 / gfx1250 (wave32, WMMA) implementation of the conv VQ-VAE forward.
// Heavy ops (all conv1d / convT1d layers + VQ distance GEMM) run on
// v_wmma_f32_16x16x32_bf16 with f32 accumulation.
// Data movement:
//   - B (im2col patch) tile: async global->LDS (ASYNCcnt + s_wait_asynccnt)
//   - A (weight) tile, direct convs: Tensor Data Mover tensor_load_to_lds
//     (TENSORcnt + s_wait_tensorcnt), one DMA per K-chunk per workgroup.
// ---------------------------------------------------------------------------

typedef __attribute__((ext_vector_type(16))) __bf16    v16bf;
typedef __attribute__((ext_vector_type(8)))  float     v8f;
typedef __attribute__((ext_vector_type(4)))  unsigned  v4u;
typedef __attribute__((ext_vector_type(8)))  unsigned  v8u;

#define BN_EPS 1e-5f

// ---------------------------------------------------------------------------
// Implicit-GEMM conv1d (direct or transposed) in bf16 WMMA, f32 accumulate.
//   direct:      y[n,co,t] = b[co] + sum_{ci,kk} w[co,ci,kk] * x[n,ci,t*s+kk-pad]
//   transposed:  y[n,co,t] = b[co] + sum_{ci,kk} w[ci,co,kk] * x[n,ci,(t+pad-kk)/s]
// GEMM view: M = Cout (16 rows/wave, 8 waves = 256 threads), N = 16 output
// positions per WG, K = Cin*ksz in chunks of 32.
// ---------------------------------------------------------------------------
__global__ __launch_bounds__(256)
void conv1d_wmma_bf16(const float* __restrict__ x,
                      const float* __restrict__ w,
                      const float* __restrict__ bias,
                      float* __restrict__ y,
                      int N, int Cin, int Cout, int Lin, int Lout,
                      int stride, int ksz, int pad, int transposed)
{
    const int tid   = threadIdx.x;
    const int lane  = tid & 31;
    const int wave  = tid >> 5;
    const int col   = lane & 15;
    const int npos0 = blockIdx.x * 16;
    const int NP    = N * Lout;
    const int K     = Cin * ksz;

    __shared__ float Bsf[16 * 32];      // B patch  [col][k] : col*32 + kl
    __shared__ float Wsf[128 * 32];     // A weights [row][k] : row*32 + kl (TDM dest)

    // ---- per-thread B-gather coordinates, fixed for the whole K loop ----
    const int bn  = tid & 15;          // which output-position column this thread feeds
    const int kl0 = tid >> 4;          // first k-row it owns (second is kl0+16)
    const int pp  = npos0 + bn;
    const bool pvalid = pp < NP;
    const int nb = pvalid ? pp / Lout : 0;    // hoisted: once per kernel, not per chunk
    const int ll = pvalid ? pp % Lout : 0;
    const float* xrow = x + (long)nb * Cin * Lin;

    // incremental (ci,kk) decomposition of kg = k0 + kl for the two owned rows
    int ciS[2], kkS[2];
    ciS[0] = kl0 / ksz;        kkS[0] = kl0 % ksz;
    ciS[1] = (kl0 + 16) / ksz; kkS[1] = (kl0 + 16) % ksz;
    const int dci = 32 / ksz, dkk = 32 % ksz;

    const int  kbase = (lane >> 4) * 8;     // ISA 16-bit A-layout K base
    const int  row   = wave * 16 + col;
    const bool rowok = row < Cout;
    const unsigned lds0 = (unsigned)(size_t)&Bsf[bn * 32 + kl0];
    const unsigned lds1 = (unsigned)(size_t)&Bsf[bn * 32 + kl0 + 16];
    const unsigned ldsW = (unsigned)(size_t)&Wsf[0];

    v8f acc = {};

    for (int k0 = 0; k0 < K; k0 += 32) {
        // ---- A tile via Tensor Data Mover (direct convs): one 2D D# DMA ----
        // tile = rows 0..Cout-1, cols k0..k0+31 of the (Cout x K) f32 weight
        // matrix; tensor_dim0 = K-k0 so the K tail is zero-filled by TDM OOB.
        if (!transposed && wave == 0) {
            unsigned long long ga = (unsigned long long)(size_t)(w + k0);
            unsigned rem = (unsigned)(K - k0);
            v4u g0;
            g0[0] = 1u;                                   // count=1 valid descriptor
            g0[1] = ldsW;                                 // lds_addr (bytes)
            g0[2] = (unsigned)ga;                         // global_addr[31:0]
            g0[3] = (unsigned)(ga >> 32) | (2u << 30);    // global_addr[56:32] | type=2
            v8u g1;
            g1[0] = 2u << 16;                             // data_size = 4 bytes
            g1[1] = (rem & 0xFFFFu) << 16;                // tensor_dim0[15:0]
            g1[2] = (rem >> 16) | (((unsigned)Cout & 0xFFFFu) << 16); // td0 hi | td1 lo
            g1[3] = 32u << 16;                            // td1 hi = 0 | tile_dim0 = 32
            g1[4] = (unsigned)Cout & 0xFFFFu;             // tile_dim1 = Cout, tile_dim2 = 0
            g1[5] = (unsigned)K;                          // tensor_dim0_stride lo32
            g1[6] = 0u;                                   // stride hi | dim1_stride lo
            g1[7] = 0u;                                   // dim1_stride hi
            asm volatile("tensor_load_to_lds %0, %1" :: "s"(g0), "s"(g1) : "memory");
        }

        // ---- B tile: zero-fill owned LDS slots, then async-DMA valid elems ----
        Bsf[bn * 32 + kl0]      = 0.f;
        Bsf[bn * 32 + kl0 + 16] = 0.f;
        asm volatile("s_wait_dscnt 0x0" ::: "memory");   // own zeros land first
        #pragma unroll
        for (int j = 0; j < 2; ++j) {
            const int kl = kl0 + 16 * j;
            const int kg = k0 + kl;
            if (kg < K && pvalid) {
                const int ci = ciS[j], kk = kkS[j];
                const float* ga = nullptr;
                if (!transposed) {
                    int ip = ll * stride + kk - pad;
                    if (ip >= 0 && ip < Lin) ga = xrow + (long)ci * Lin + ip;
                } else {
                    int num = ll + pad - kk;
                    if (num >= 0 && (num % stride) == 0) {
                        int ip = num / stride;
                        if (ip < Lin) ga = xrow + (long)ci * Lin + ip;
                    }
                }
                if (ga) {
                    unsigned loff = j ? lds1 : lds0;
                    asm volatile("global_load_async_to_lds_b32 %0, %1, off"
                                 :: "v"(loff),
                                    "v"((unsigned long long)(size_t)ga)
                                 : "memory");
                }
            }
            // advance (ci,kk) by 32 K-steps for the next chunk (no div/mod)
            kkS[j] += dkk; ciS[j] += dci;
            if (kkS[j] >= ksz) { kkS[j] -= ksz; ciS[j] += 1; }
        }
        asm volatile("s_wait_asynccnt 0x0" ::: "memory"); // own async DMAs done
        if (!transposed && wave == 0)
            __builtin_amdgcn_s_wait_tensorcnt(0);         // weight tile DMA done
        __syncthreads();                                  // tiles visible WG-wide

        // ---- A fragment: 16(M) x 32(K) weights in bf16 (ISA A layout) ----
        v16bf a, b;
        if (!transposed) {
            const float* wr = &Wsf[row * 32];             // contiguous LDS reads
            #pragma unroll
            for (int seg = 0; seg < 2; ++seg) {
                const int klb = kbase + seg * 16;
                #pragma unroll
                for (int i = 0; i < 8; ++i)
                    a[seg * 8 + i] = (__bf16)(rowok ? wr[klb + i] : 0.f);
            }
        } else {
            #pragma unroll
            for (int seg = 0; seg < 2; ++seg) {
                const int klb = kbase + seg * 16;
                int kg = k0 + klb;
                int ci = kg / ksz, kk = kg % ksz;
                #pragma unroll
                for (int i = 0; i < 8; ++i) {
                    a[seg * 8 + i] = (__bf16)((rowok && kg < K)
                                   ? w[((long)ci * Cout + row) * ksz + kk] : 0.f);
                    ++kg;
                    if (++kk == ksz) { kk = 0; ++ci; }
                }
            }
            if (k0 + 32 < K)   // hint next chunk toward GL2 (global_prefetch_b8)
                __builtin_prefetch(w + (long)(rowok ? row : 0) * ksz, 0, 1);
        }
        // ---- B fragment: two runs of 8 floats per lane (k = 8h+i, 16+8h+i) ----
        {
            const float* bp = &Bsf[col * 32 + ((lane < 16) ? 0 : 8)];
            #pragma unroll
            for (int i = 0; i < 16; ++i) b[i] = (__bf16)bp[(i < 8) ? i : (i + 8)];
        }

        acc = __builtin_amdgcn_wmma_f32_16x16x32_bf16(false, a, false, b,
                                                      (short)0, acc, false, false);
        __syncthreads();
    }

    // ---- epilogue: C layout VGPR r -> M = r (lanes 0-15) / r+8 (lanes 16-31)
    if (pvalid) {
        #pragma unroll
        for (int r = 0; r < 8; ++r) {
            const int m  = (lane < 16) ? r : (r + 8);
            const int co = wave * 16 + m;
            if (co < Cout)
                y[((long)nb * Cout + co) * Lout + ll] = acc[r] + bias[co];
        }
    }
}

// ---------------------------------------------------------------------------
// BatchNorm (training-mode batch stats): per-channel mean + rsqrt(var+eps)
// ---------------------------------------------------------------------------
__global__ void bn_stats(const float* __restrict__ x, float* __restrict__ mean,
                         float* __restrict__ inv, int N, int C, int L)
{
    int c = blockIdx.x;
    long total = (long)N * L;
    float s = 0.f, s2 = 0.f;
    for (long i = threadIdx.x; i < total; i += blockDim.x) {
        int n = (int)(i / L), l = (int)(i % L);
        float v = x[((long)n * C + c) * L + l];
        s += v; s2 += v * v;
    }
    __shared__ float ss[256], ss2[256];
    ss[threadIdx.x] = s; ss2[threadIdx.x] = s2;
    __syncthreads();
    for (int o = blockDim.x >> 1; o > 0; o >>= 1) {
        if ((int)threadIdx.x < o) {
            ss[threadIdx.x]  += ss[threadIdx.x + o];
            ss2[threadIdx.x] += ss2[threadIdx.x + o];
        }
        __syncthreads();
    }
    if (threadIdx.x == 0) {
        float m   = ss[0] / (float)total;
        float var = ss2[0] / (float)total - m * m;
        mean[c] = m;
        inv[c]  = rsqrtf(var + BN_EPS);
    }
}

__global__ void bn_relu_inplace(float* __restrict__ x, const float* __restrict__ mean,
                                const float* __restrict__ inv, const float* __restrict__ g,
                                const float* __restrict__ be, int N, int C, int L)
{
    long i = (long)blockIdx.x * blockDim.x + threadIdx.x;
    long total = (long)N * C * L;
    if (i < total) {
        int c = (int)((i / L) % C);
        float v = (x[i] - mean[c]) * inv[c] * g[c] + be[c];
        x[i] = v > 0.f ? v : 0.f;
    }
}

__global__ void bn_add(const float* __restrict__ res, const float* __restrict__ h,
                       const float* __restrict__ mean, const float* __restrict__ inv,
                       const float* __restrict__ g, const float* __restrict__ be,
                       float* __restrict__ out, int N, int C, int L)
{
    long i = (long)blockIdx.x * blockDim.x + threadIdx.x;
    long total = (long)N * C * L;
    if (i < total) {
        int c = (int)((i / L) % C);
        out[i] = res[i] + (h[i] - mean[c]) * inv[c] * g[c] + be[c];
    }
}

// ---------------------------------------------------------------------------
// (N,C,S) <-> row-major (N*S, C) layout shuffles for the VQ stage
// ---------------------------------------------------------------------------
__global__ void nchw_to_rows(const float* __restrict__ h, float* __restrict__ rows,
                             int N, int C, int S)
{
    long i = (long)blockIdx.x * blockDim.x + threadIdx.x;
    long total = (long)N * C * S;
    if (i < total) {
        int d = (int)(i % C);
        int t = (int)((i / C) % S);
        int n = (int)(i / ((long)C * S));
        rows[i] = h[((long)n * C + d) * S + t];
    }
}

__global__ void rows_to_nchw(const float* __restrict__ rows, float* __restrict__ h,
                             int N, int C, int S)
{
    long i = (long)blockIdx.x * blockDim.x + threadIdx.x;
    long total = (long)N * C * S;
    if (i < total) {
        int d = (int)(i % C);
        int t = (int)((i / C) % S);
        int n = (int)(i / ((long)C * S));
        h[((long)n * C + d) * S + t] = rows[i];
    }
}

__global__ void cb_norms(const float* __restrict__ cb, float* __restrict__ wn,
                         int total, int D)
{
    int j = blockIdx.x * blockDim.x + threadIdx.x;
    if (j < total) {
        float s = 0.f;
        for (int d = 0; d < D; ++d) { float v = cb[(long)j * D + d]; s += v * v; }
        wn[j] = s;
    }
}

// ---------------------------------------------------------------------------
// VQ nearest-codeword search: res (R x 64) against W (1024 x 64).
// One wave per 16-row tile; A fragments (res rows) are loop-invariant and
// hoisted; dot products via two bf16 WMMA (K=64); argmin of ||w||^2 - 2*dot
// via wave32 shfl_xor butterfly (width 16, matching the C lane layout).
// ---------------------------------------------------------------------------
__global__ void vq_argmin_wmma(const float* __restrict__ res, const float* __restrict__ W,
                               const float* __restrict__ wn, int* __restrict__ idx, int R)
{
    const int lane = threadIdx.x & 31;
    const int col  = lane & 15;
    const int r0   = blockIdx.x * 16;
    const int row  = r0 + col;
    const int kbase = (lane >> 4) * 8;

    // hoisted A fragments (rows of res), K = 0..31 and 32..63
    v16bf a0, a1;
    #pragma unroll
    for (int seg = 0; seg < 2; ++seg) {
        const int klb = kbase + seg * 16;
        #pragma unroll
        for (int i = 0; i < 8; ++i) {
            int kl = klb + i;
            a0[seg * 8 + i] = (__bf16)((row < R) ? res[(long)row * 64 + kl]      : 0.f);
            a1[seg * 8 + i] = (__bf16)((row < R) ? res[(long)row * 64 + 32 + kl] : 0.f);
        }
    }

    float best[8];
    int   bestj[8];
    #pragma unroll
    for (int r = 0; r < 8; ++r) { best[r] = 3.4e38f; bestj[r] = 0; }

    for (int j0 = 0; j0 < 1024; j0 += 16) {
        const int jc = j0 + col;
        v16bf b0, b1;
        {
            const int klb = (lane < 16) ? 0 : 8;
            const float* wp = W + (long)jc * 64 + klb;
            #pragma unroll
            for (int i = 0; i < 16; ++i) {
                const int ki = (i < 8) ? i : (i + 8);
                b0[i] = (__bf16)wp[ki];
                b1[i] = (__bf16)wp[32 + ki];
            }
        }
        v8f acc = {};
        acc = __builtin_amdgcn_wmma_f32_16x16x32_bf16(false, a0, false, b0,
                                                      (short)0, acc, false, false);
        acc = __builtin_amdgcn_wmma_f32_16x16x32_bf16(false, a1, false, b1,
                                                      (short)0, acc, false, false);
        // distance (up to the row-constant ||res||^2) and butterfly argmin
        #pragma unroll
        for (int r = 0; r < 8; ++r) {
            float v  = wn[jc] - 2.0f * acc[r];
            int   bj = jc;
            #pragma unroll
            for (int off = 1; off < 16; off <<= 1) {
                float ov = __shfl_xor(v,  off, 16);
                int   oj = __shfl_xor(bj, off, 16);
                if (ov < v) { v = ov; bj = oj; }
            }
            if (v < best[r]) { best[r] = v; bestj[r] = bj; }
        }
    }
    if (col == 0) {
        #pragma unroll
        for (int r = 0; r < 8; ++r) {
            const int m  = (lane < 16) ? r : (r + 8);
            const int rr = r0 + m;
            if (rr < R) idx[rr] = bestj[r];
        }
    }
}

// res -= near; q += near; sse += (res_pre - near)^2; record code (as float)
__global__ void vq_update(float* __restrict__ res, float* __restrict__ q,
                          const float* __restrict__ W, const int* __restrict__ idx,
                          float* __restrict__ sse, float* __restrict__ disc,
                          int R, int D, int cb_i, int ncb)
{
    long i = (long)blockIdx.x * blockDim.x + threadIdx.x;
    long total = (long)R * D;
    if (i < total) {
        int row = (int)(i / D), d = (int)(i % D);
        int j = idx[row];
        float nv = W[(long)j * D + d];
        float rv = res[i];
        float df = rv - nv;
        atomicAdd(sse, df * df);
        res[i] = rv - nv;
        q[i]  += nv;
        if (d == 0) disc[(long)row * ncb + cb_i] = (float)j;
    }
}

__global__ void zero_f32(float* __restrict__ p, long n)
{
    long i = (long)blockIdx.x * blockDim.x + threadIdx.x;
    if (i < n) p[i] = 0.f;
}

__global__ void finalize_losses(const float* __restrict__ sse, float* __restrict__ out,
                                float scale)
{
    if (threadIdx.x == 0) {
        float v = sse[0] * scale;   // loss_cd == loss_enc in forward value
        out[0] = v;
        out[1] = v;
    }
}

// ---------------------------------------------------------------------------
// Host orchestration
// ---------------------------------------------------------------------------
static void launch_conv(hipStream_t st, const float* x, const float* w, const float* b,
                        float* y, int N, int Cin, int Cout, int Lin, int Lout,
                        int stride, int k, int pad, int transposed)
{
    dim3 blk(256);                                   // always 8 wave32s
    dim3 grd((unsigned)((N * Lout + 15) / 16));
    conv1d_wmma_bf16<<<grd, blk, 0, st>>>(x, w, b, y, N, Cin, Cout, Lin, Lout,
                                          stride, k, pad, transposed);
}

static inline unsigned egrid(long n) { return (unsigned)((n + 255) / 256); }

extern "C" void kernel_launch(void* const* d_in, const int* in_sizes, int n_in,
                              void* d_out, int out_size, void* d_ws, size_t ws_size,
                              hipStream_t stream)
{
    (void)in_sizes; (void)n_in; (void)out_size; (void)ws_size;
    const int N = 8;

    const float* x = (const float*)d_in[0];
    auto P = [&](int i) { return (const float*)d_in[i]; };
    // param order per setup_inputs insertion order:
    //   [0]=x, down blocks at 1+10*i, up blocks at 41+10*i, cb at 81
    //   per block: +0 w1 +1 b1 +2 g1 +3 be1 +4 w2 +5 b2 +6 g2 +7 be2 +8 rw +9 rb
    const float* cb = P(81);

    // ---- workspace carve ----
    char* ws = (char*)d_ws;
    float* T1 = (float*)(ws);                         // conv1 out (max 8*128*128000 f32)
    float* T2 = (float*)(ws +  524288000ull);         // conv2 out (max 131 MB)
    float* T3 = (float*)(ws +  655360000ull);         // residual-conv out
    float* A0 = (float*)(ws +  786432000ull);         // activation ping
    float* A1 = (float*)(ws +  917504000ull);         // activation pong
    char*  SM = ws + 1048576000ull;                   // small scratch
    float* MEAN = (float*)SM;                         // 128
    float* INVS = MEAN + 128;                         // 128
    float* WN   = INVS + 128;                         // 8*1024
    float* RES  = WN + 8192;                          // 1600*64
    float* Q    = RES + 102400;                       // 1600*64
    int*   IDX  = (int*)(Q + 102400);                 // 8*1600
    float* SSE  = (float*)(IDX + 12800);              // 1

    // ---- model configuration ----
    const int dcf[4][4] = {{1,128,4,7},{128,128,4,7},{128,128,5,7},{128,64,8,3}};
    const int ucf[4][5] = {{64,128,8,3,0},{128,128,5,7,1},{128,128,4,7,2},{128,1,4,7,3}};

    // final output length (for d_out layout: h | loss_cd | loss_enc | discrete)
    int Lf = 128000;
    for (int i = 0; i < 4; ++i)
        Lf = (Lf + 2 * (dcf[i][3] / 2) - dcf[i][3]) / dcf[i][2] + 1;         // -> 200
    const int Svq = Lf;                                                       // 200
    for (int i = 0; i < 4; ++i)
        Lf = (Lf - 1) * ucf[i][2] + ucf[i][3] - 2 * (ucf[i][3] / 2) + ucf[i][4];
    const long hN = (long)N * Lf;
    float* out_h    = (float*)d_out;
    float* out_loss = out_h + hN;
    float* out_disc = out_loss + 2;

    // ================= encoder (down path) =================
    const float* cur = x;
    float* nxt = A0;
    int Lin = 128000;
    for (int i = 0; i < 4; ++i) {
        int cin = dcf[i][0], cout = dcf[i][1], s = dcf[i][2], k = dcf[i][3];
        int base = 1 + 10 * i;
        int pad  = k / 2;
        int Lout = (Lin + 2 * pad - k) / s + 1;

        launch_conv(stream, cur, P(base+0), P(base+1), T1, N, cin, cout, Lin, Lin, 1, k, pad, 0);
        bn_stats<<<cout, 256, 0, stream>>>(T1, MEAN, INVS, N, cout, Lin);
        bn_relu_inplace<<<egrid((long)N*cout*Lin), 256, 0, stream>>>(T1, MEAN, INVS,
                                                                     P(base+2), P(base+3),
                                                                     N, cout, Lin);
        launch_conv(stream, T1, P(base+4), P(base+5), T2, N, cout, cout, Lin, Lout, s, k, pad, 0);
        bn_stats<<<cout, 256, 0, stream>>>(T2, MEAN, INVS, N, cout, Lout);
        launch_conv(stream, cur, P(base+8), P(base+9), T3, N, cin, cout, Lin, Lout, s, 1, 0, 0);
        bn_add<<<egrid((long)N*cout*Lout), 256, 0, stream>>>(T3, T2, MEAN, INVS,
                                                             P(base+6), P(base+7),
                                                             nxt, N, cout, Lout);
        cur = nxt;
        nxt = (nxt == A0) ? A1 : A0;
        Lin = Lout;
    }
    // cur = enc (N, 64, Svq)

    // ================= residual VQ =================
    const int R = N * Svq, D = 64;
    zero_f32<<<1, 32, 0, stream>>>(SSE, 1);
    zero_f32<<<egrid((long)R * D), 256, 0, stream>>>(Q, (long)R * D);
    cb_norms<<<egrid(8 * 1024), 256, 0, stream>>>(cb, WN, 8 * 1024, D);
    nchw_to_rows<<<egrid((long)R * D), 256, 0, stream>>>(cur, RES, N, D, Svq);
    for (int i = 0; i < 8; ++i) {
        const float* Wi = cb + (long)i * 1024 * D;
        vq_argmin_wmma<<<(R + 15) / 16, 32, 0, stream>>>(RES, Wi, WN + i * 1024,
                                                         IDX + (long)i * R, R);
        vq_update<<<egrid((long)R * D), 256, 0, stream>>>(RES, Q, Wi, IDX + (long)i * R,
                                                          SSE, out_disc, R, D, i, 8);
    }
    finalize_losses<<<1, 32, 0, stream>>>(SSE, out_loss, 1.0f / (float)((long)R * D));
    rows_to_nchw<<<egrid((long)R * D), 256, 0, stream>>>(Q, nxt, N, D, Svq);  // quant == q
    cur = nxt;
    nxt = (nxt == A0) ? A1 : A0;

    // ================= decoder (up path) =================
    Lin = Svq;
    for (int i = 0; i < 4; ++i) {
        int cin = ucf[i][0], cout = ucf[i][1], s = ucf[i][2], k = ucf[i][3], op = ucf[i][4];
        int base = 41 + 10 * i;
        int pad  = k / 2;
        int Lout = (Lin - 1) * s + k - 2 * pad + op;

        launch_conv(stream, cur, P(base+0), P(base+1), T1, N, cin, cout, Lin, Lin, 1, k, pad, 1);
        bn_stats<<<cout, 256, 0, stream>>>(T1, MEAN, INVS, N, cout, Lin);
        bn_relu_inplace<<<egrid((long)N*cout*Lin), 256, 0, stream>>>(T1, MEAN, INVS,
                                                                     P(base+2), P(base+3),
                                                                     N, cout, Lin);
        launch_conv(stream, T1, P(base+4), P(base+5), T2, N, cout, cout, Lin, Lout, s, k, pad, 1);
        bn_stats<<<cout, 256, 0, stream>>>(T2, MEAN, INVS, N, cout, Lout);
        launch_conv(stream, cur, P(base+8), P(base+9), T3, N, cin, cout, Lin, Lout, s, 1, 0, 1);
        float* dst = (i == 3) ? out_h : nxt;
        bn_add<<<egrid((long)N*cout*Lout), 256, 0, stream>>>(T3, T2, MEAN, INVS,
                                                             P(base+6), P(base+7),
                                                             dst, N, cout, Lout);
        cur = dst;
        nxt = (nxt == A0) ? A1 : A0;
        Lin = Lout;
    }
}
